// SIM_38654705664505
// MI455X (gfx1250) — matrix operand order repaired and checked by reading, not compiled, not run on hardware
//
#include <hip/hip_runtime.h>

typedef __attribute__((ext_vector_type(16))) __bf16          v16bf;
typedef __attribute__((ext_vector_type(16))) unsigned short  v16u;
typedef __attribute__((ext_vector_type(8)))  unsigned short  v8u;
typedef __attribute__((ext_vector_type(8)))  float           v8f;

#define HW_   9216
#define CEMB  256
#define KP    576      // 24*24 pooled positions
#define CH1   288      // 264 concat channels padded to 9*32
#define K1    2592     // 9 taps * 288
#define K2    1152     // 9 taps * 128

// ---------- CDNA5 async / TDM availability ----------
#if __has_builtin(__builtin_amdgcn_global_load_async_to_lds_b128) && \
    __has_builtin(__builtin_amdgcn_s_wait_asynccnt)
#define HAVE_ASYNC_LDS 1
#else
#define HAVE_ASYNC_LDS 0
#endif
#if __has_builtin(__builtin_amdgcn_tensor_load_to_lds) && \
    __has_builtin(__builtin_amdgcn_s_wait_tensorcnt)
#define HAVE_TDM 1
#else
#define HAVE_TDM 0
#endif

#if HAVE_ASYNC_LDS
// ROCm 7.2 clang-22 signature (from diagnostic): parameters are generic
// pointers to 'int __vector(4)' (int4*), i.e. (src_global, dst_lds, imm, imm).
typedef int a_i32x4 __attribute__((vector_size(16)));
__device__ __forceinline__ void async_b128(const unsigned short* g, unsigned short* l) {
  __builtin_amdgcn_global_load_async_to_lds_b128(
      (a_i32x4*)(void*)g, (a_i32x4*)(void*)l, 0, 0);
}
#endif

#if HAVE_TDM
typedef __attribute__((ext_vector_type(4))) unsigned int tdm_u32x4;
typedef __attribute__((ext_vector_type(8))) int          tdm_i32x8;
typedef __attribute__((ext_vector_type(4))) int          tdm_i32x4;
// 3D TDM tile: dim0 = 32 ch (stride 1), dim1 = 16 x-pixels (stride 256),
// dim2 = 16 y-pixels (stride 96*256); data_size=2B. LDS pad: every 16 DWORDs
// (64B = 32ch) insert 4 DWORDs (16B) -> reproduces the ushort-stride-40 layout.
__device__ __forceinline__ void tdm_halo_load(const unsigned short* gsrc, unsigned int lds_addr) {
  unsigned long long ga = (unsigned long long)(size_t)gsrc;
  tdm_u32x4 g0;
  g0[0] = 1u;                                   // D# count = 1 (valid user descriptor)
  g0[1] = lds_addr;                             // lds_addr [63:32]
  g0[2] = (unsigned int)ga;                     // global_addr [95:64]
  g0[3] = (unsigned int)((ga >> 32) & 0x01FFFFFFull) | 0x80000000u;  // addr[56:32] | type=2
  tdm_i32x8 g1;
  g1[0] = (int)((1u << 16) | (1u << 20) | (3u << 22) | (3u << 25));
  //            data_size=2B  pad_enable  pad_interval=16dw  pad_amount=4dw
  g1[1] = (int)(32u << 16);                     // tensor_dim0 = 32  (bits 79:48)
  g1[2] = (int)(16u << 16);                     // tensor_dim1 = 16  (bits 111:80)
  g1[3] = (int)(32u << 16);                     // tile_dim0   = 32  (bits 127:112)
  g1[4] = (int)(16u | (16u << 16));             // tile_dim1=16, tile_dim2=16
  g1[5] = 256;                                  // tensor_dim0_stride low (channels per pixel)
  g1[6] = (int)(24576u << 16);                  // tensor_dim1_stride low16 (96*256) at bit 208
  g1[7] = 0;
  tdm_i32x4 g2;
  g2[0] = 16; g2[1] = 0; g2[2] = 0; g2[3] = 0;  // tensor_dim2 = 16
  tdm_i32x4 g3;
  g3[0] = 0; g3[1] = 0; g3[2] = 0; g3[3] = 0;
#if __clang_major__ >= 23
  tdm_i32x8 g4 = {0, 0, 0, 0, 0, 0, 0, 0};
  __builtin_amdgcn_tensor_load_to_lds(g0, g1, g2, g3, g4, 0);
#else
  __builtin_amdgcn_tensor_load_to_lds(g0, g1, g2, g3, 0);
#endif
}
#endif

// ---------- bf16 helpers (ushort-based, RNE) ----------
__device__ __forceinline__ unsigned short f2bf(float f) {
  unsigned u = __float_as_uint(f);
  unsigned r = u + 0x7FFFu + ((u >> 16) & 1u);
  return (unsigned short)(r >> 16);
}
__device__ __forceinline__ float bf2f(unsigned short h) {
  return __uint_as_float(((unsigned)h) << 16);
}
__device__ __forceinline__ v8f wmma_bf16(v16u a, v16u b, v8f c) {
  return __builtin_amdgcn_wmma_f32_16x16x32_bf16(
      false, __builtin_bit_cast(v16bf, a),
      false, __builtin_bit_cast(v16bf, b),
      (short)0, c, false, false);
}
// A-fragment: lane holds row M=lane&15; hi = lane>>4 selects K sub-ranges
// VGPR0-7: K = k0 + hi*16 .. +15 (contiguous, matches B fragments)
__device__ __forceinline__ v16u load_afrag(const unsigned short* row, int k0, int hi) {
  union { v16u v; v8u h[2]; } u;
  u.h[0] = *(const v8u*)(row + k0 + hi * 16);
  u.h[1] = *(const v8u*)(row + k0 + hi * 16 + 8);
  return u.v;
}

// ---------- 1. avg_pool4(init_embed) -> fi_t bf16 [b][k=576][c=256] ----------
__global__ __launch_bounds__(256) void k_pool_fi(const float* __restrict__ init_embed,
                                                 unsigned short* __restrict__ fi_t) {
  int tid = blockIdx.x * 256 + threadIdx.x;          // 8*256*576 threads
  int k = tid % KP;
  int t = tid / KP;
  int c = t & 255;
  int b = t >> 8;
  int y = k / 24, x = k % 24;
  const float* base = init_embed + (((size_t)(b * CEMB + c) * 96 + y * 4) * 96 + x * 4);
  float s = 0.f;
#pragma unroll
  for (int r = 0; r < 4; ++r) {
    float4 v = *(const float4*)(base + r * 96);
    s += v.x + v.y + v.z + v.w;
  }
  fi_t[((size_t)b * KP + k) * CEMB + c] = f2bf(s * (1.f / 16.f));
}

// ---------- 2. avg_pool4(init_seg) -> si_t f32 [b][k][2] ----------
__global__ __launch_bounds__(256) void k_pool_si(const float* __restrict__ init_seg,
                                                 float* __restrict__ si_t) {
  int tid = blockIdx.x * 256 + threadIdx.x;          // 8*576 threads
  int k = tid % KP;
  int b = tid / KP;
  int y = k / 24, x = k % 24;
#pragma unroll
  for (int cls = 0; cls < 2; ++cls) {
    const float* base = init_seg + (((size_t)(b * 2 + cls) * 96 + y * 4) * 96 + x * 4);
    float s = 0.f;
#pragma unroll
    for (int r = 0; r < 4; ++r) {
      float4 v = *(const float4*)(base + r * 96);
      s += v.x + v.y + v.z + v.w;
    }
    si_t[((size_t)b * KP + k) * 2 + cls] = s * (1.f / 16.f);
  }
}

// ---------- 3. transpose fp32 [b][256][9216] -> bf16 [b][pix][256] ----------
__global__ __launch_bounds__(256) void k_transpose(const float* __restrict__ src,
                                                   unsigned short* __restrict__ dst) {
  int gid = blockIdx.x * 256 + threadIdx.x;          // 8*9216 threads
  int b = gid / HW_;
  int p = gid % HW_;
  const float* s = src + (size_t)b * CEMB * HW_ + p;
  unsigned short* d = dst + ((size_t)b * HW_ + p) * CEMB;
  for (int c0 = 0; c0 < CEMB; c0 += 8) {
    union { v8u v; unsigned short u[8]; } pk;
#pragma unroll
    for (int j = 0; j < 8; ++j) pk.u[j] = f2bf(s[(size_t)(c0 + j) * HW_]);
    *(v8u*)(d + c0) = pk.v;
  }
}

// ---------- 4. weight prep: W1 (128,264,3,3) -> W1s bf16 [m][K1] (k = tap*288 + c) ----------
__global__ __launch_bounds__(256) void k_w1prep(const float* __restrict__ W1,
                                                unsigned short* __restrict__ W1s) {
  int gid = blockIdx.x * 256 + threadIdx.x;          // 128*2592 threads
  int m = gid / K1;
  int k = gid % K1;
  int tap = k / CH1, c = k % CH1;
  unsigned short v = 0;
  if (c < 264) {
    int ky = tap / 3, kx = tap % 3;
    v = f2bf(W1[(((size_t)m * 264 + c) * 3 + ky) * 3 + kx]);
  }
  W1s[gid] = v;
}

// ---------- 5. weight prep: W2 (2,128,3,3) -> W2s bf16 [16][K2] (rows 2..15 zero) ----------
__global__ __launch_bounds__(256) void k_w2prep(const float* __restrict__ W2,
                                                unsigned short* __restrict__ W2s) {
  int gid = blockIdx.x * 256 + threadIdx.x;          // 16*1152 threads
  int m = gid / K2;
  int k = gid % K2;
  int tap = k / 128, c = k % 128;
  unsigned short v = 0;
  if (m < 2) {
    int ky = tap / 3, kx = tap % 3;
    v = f2bf(W2[(((size_t)m * 128 + c) * 3 + ky) * 3 + kx]);
  }
  W2s[gid] = v;
}

// ---------- 6. gmap: WMMA sim = fi(576x256) x curT(256xN), weighted max over 576 ----------
__global__ __launch_bounds__(256) void k_gmap(const unsigned short* __restrict__ fi_t,
                                              const float* __restrict__ si_t,
                                              const unsigned short* __restrict__ curT,
                                              float* __restrict__ gmap) {
  __shared__ float siL[KP * 2];
  int b = blockIdx.x / 72;
  for (int i = threadIdx.x; i < KP * 2; i += 256) siL[i] = si_t[(size_t)b * KP * 2 + i];
  __syncthreads();
  int wave = threadIdx.x >> 5, lane = threadIdx.x & 31;
  int hi = lane >> 4, l = lane & 15;
  int n = (blockIdx.x % 72) * 128 + wave * 16 + l;   // pixel column for this lane
  const unsigned short* colp = curT + ((size_t)b * HW_ + n) * CEMB + hi * 16;
  v16u bfr[8];
#pragma unroll
  for (int f = 0; f < 8; ++f) bfr[f] = *(const v16u*)(colp + f * 32);
  float b0 = -3.0e38f, b1 = -3.0e38f;
  for (int mt = 0; mt < 36; ++mt) {
    v8f acc = {};
    const unsigned short* arow = fi_t + ((size_t)b * KP + mt * 16 + l) * CEMB;
#pragma unroll
    for (int f = 0; f < 8; ++f)
      acc = wmma_bf16(load_afrag(arow, f * 32, hi), bfr[f], acc);
#pragma unroll
    for (int i = 0; i < 8; ++i) {
      int m = mt * 16 + hi * 8 + i;
      b0 = fmaxf(b0, acc[i] * siL[m * 2 + 0]);
      b1 = fmaxf(b1, acc[i] * siL[m * 2 + 1]);
    }
  }
  b0 = fmaxf(b0, __shfl_xor(b0, 16));
  b1 = fmaxf(b1, __shfl_xor(b1, 16));
  if (hi == 0) {
    gmap[((size_t)b * 2 + 0) * HW_ + n] = b0;
    gmap[((size_t)b * 2 + 1) * HW_ + n] = b1;
  }
}

// ---------- 7. local correlation: 8x8 tile, 16x16 halo, channel-chunked LDS ----------
__global__ __launch_bounds__(256) void k_corr(const unsigned short* __restrict__ curT,
                                              const unsigned short* __restrict__ prevT,
                                              const float* __restrict__ prev_seg,
                                              float* __restrict__ lmap) {
  __shared__ unsigned short prevL[256 * 40];   // 16x16 halo x 32ch (ushort stride 40)
  __shared__ unsigned short curL[64 * 40];     // 8x8 tile x 32ch
  __shared__ float rAcc[64 * 81];              // raw dot per (pixel, displacement)
  __shared__ float segL[256 * 2];              // seg halo
  int tid = threadIdx.x;
  int blk = blockIdx.x;
  int b = blk / 144;
  int t = blk % 144;
  int oy = (t / 12) * 8, ox = (t % 12) * 8;
  int hy0 = oy - 4, hx0 = ox - 4;
  bool interior = (hy0 >= 0) && (hx0 >= 0) && (hy0 + 16 <= 96) && (hx0 + 16 <= 96);
  (void)interior;
  for (int i = tid; i < 64 * 81; i += 256) rAcc[i] = 0.f;
  {
    int hy = hy0 + tid / 16, hx = hx0 + tid % 16;
    bool v = ((unsigned)hy < 96u) && ((unsigned)hx < 96u);
    int cy = v ? hy : 0, cx = v ? hx : 0;
    segL[tid * 2 + 0] = v ? prev_seg[((size_t)b * 2 + 0) * HW_ + cy * 96 + cx] : 0.f;
    segL[tid * 2 + 1] = v ? prev_seg[((size_t)b * 2 + 1) * HW_ + cy * 96 + cx] : 0.f;
  }
  int pixel = tid >> 2, part = tid & 3;
  int py = pixel >> 3, px = pixel & 7;
  for (int c0 = 0; c0 < CEMB; c0 += 32) {
    __syncthreads();
    bool halo_done = false;
#if HAVE_TDM
    if (interior) {   // whole halo tile via Tensor Data Mover (one op per chunk)
      if (tid < 32) {
        tdm_halo_load(prevT + (((size_t)b * HW_ + hy0 * 96 + hx0) * CEMB + c0),
                      (unsigned int)(size_t)&prevL[0]);
        __builtin_amdgcn_s_wait_tensorcnt(0);
      }
      halo_done = true;
    }
#endif
    if (!halo_done) {  // edge tiles: per-lane path with OOB zeroing
      int hy = hy0 + tid / 16, hx = hx0 + tid % 16;
      bool v = ((unsigned)hy < 96u) && ((unsigned)hx < 96u);
      int cy = v ? hy : 0, cx = v ? hx : 0;
      const unsigned short* sp = prevT + ((size_t)b * HW_ + cy * 96 + cx) * CEMB + c0;
#if HAVE_ASYNC_LDS
      if (v) {
#pragma unroll
        for (int j = 0; j < 4; ++j) async_b128(sp + j * 8, &prevL[tid * 40 + j * 8]);
      } else {
        v8u zz = {};
#pragma unroll
        for (int j = 0; j < 4; ++j) *(v8u*)(&prevL[tid * 40 + j * 8]) = zz;
      }
#else
#pragma unroll
      for (int j = 0; j < 4; ++j) {
        v8u vv = {};
        if (v) vv = *(const v8u*)(sp + j * 8);
        *(v8u*)(&prevL[tid * 40 + j * 8]) = vv;
      }
#endif
    }
    if (tid < 64) {  // cur tile (always in range)
      int cy = oy + (tid >> 3), cx = ox + (tid & 7);
      const unsigned short* sp = curT + ((size_t)b * HW_ + cy * 96 + cx) * CEMB + c0;
#if HAVE_ASYNC_LDS
#pragma unroll
      for (int j = 0; j < 4; ++j) async_b128(sp + j * 8, &curL[tid * 40 + j * 8]);
#else
#pragma unroll
      for (int j = 0; j < 4; ++j) *(v8u*)(&curL[tid * 40 + j * 8]) = *(const v8u*)(sp + j * 8);
#endif
    }
#if HAVE_ASYNC_LDS
    __builtin_amdgcn_s_wait_asynccnt(0);
#endif
    __syncthreads();
    const unsigned short* cl = &curL[pixel * 40];
    for (int d = part; d < 81; d += 4) {
      int dy = d / 9, dx = d - dy * 9;
      const unsigned short* pl = &prevL[((py + dy) * 16 + px + dx) * 40];
      float s = 0.f;
#pragma unroll
      for (int c = 0; c < 32; ++c) s += bf2f(cl[c]) * bf2f(pl[c]);
      rAcc[pixel * 81 + d] += s;
    }
  }
  __syncthreads();
  float b0 = -3.0e38f, b1 = -3.0e38f;
  for (int d = part; d < 81; d += 4) {
    int dy = d / 9, dx = d - dy * 9;
    int hp = (py + dy) * 16 + px + dx;
    float rv = rAcc[pixel * 81 + d] * (1.f / 256.f);
    b0 = fmaxf(b0, rv * segL[hp * 2 + 0]);
    b1 = fmaxf(b1, rv * segL[hp * 2 + 1]);
  }
  b0 = fmaxf(b0, __shfl_xor(b0, 1)); b0 = fmaxf(b0, __shfl_xor(b0, 2));
  b1 = fmaxf(b1, __shfl_xor(b1, 1)); b1 = fmaxf(b1, __shfl_xor(b1, 2));
  if (part == 0) {
    int gp = (oy + py) * 96 + ox + px;
    lmap[((size_t)b * 2 + 0) * HW_ + gp] = b0;
    lmap[((size_t)b * 2 + 1) * HW_ + gp] = b1;
  }
}

// ---------- 8. concat -> X bf16 [b][pix][288] (264 real + 24 zero pad) ----------
__global__ __launch_bounds__(256) void k_stageX(const float* __restrict__ cur_decode,
                                                const float* __restrict__ gmap,
                                                const float* __restrict__ init_seg,
                                                const float* __restrict__ lmap,
                                                const float* __restrict__ prev_seg,
                                                unsigned short* __restrict__ X) {
  int gid = blockIdx.x * 256 + threadIdx.x;          // 8*9216 threads
  int b = gid / HW_;
  int p = gid % HW_;
  unsigned short* xr = X + ((size_t)b * HW_ + p) * CH1;
  const float* cd = cur_decode + (size_t)b * 256 * HW_ + p;
  for (int c0 = 0; c0 < 256; c0 += 8) {
    union { v8u v; unsigned short u[8]; } pk;
#pragma unroll
    for (int j = 0; j < 8; ++j) pk.u[j] = f2bf(cd[(size_t)(c0 + j) * HW_]);
    *(v8u*)(xr + c0) = pk.v;
  }
  union { v8u v; unsigned short u[8]; } ex;
  ex.u[0] = f2bf(gmap[((size_t)b * 2 + 0) * HW_ + p]);
  ex.u[1] = f2bf(gmap[((size_t)b * 2 + 1) * HW_ + p]);
  ex.u[2] = f2bf(init_seg[((size_t)b * 2 + 0) * HW_ + p]);
  ex.u[3] = f2bf(init_seg[((size_t)b * 2 + 1) * HW_ + p]);
  ex.u[4] = f2bf(lmap[((size_t)b * 2 + 0) * HW_ + p]);
  ex.u[5] = f2bf(lmap[((size_t)b * 2 + 1) * HW_ + p]);
  ex.u[6] = f2bf(prev_seg[((size_t)b * 2 + 0) * HW_ + p]);
  ex.u[7] = f2bf(prev_seg[((size_t)b * 2 + 1) * HW_ + p]);
  *(v8u*)(xr + 256) = ex.v;
  v8u z = {};
  *(v8u*)(xr + 264) = z;
  *(v8u*)(xr + 272) = z;
  *(v8u*)(xr + 280) = z;
}

// ---------- 9. conv1: 264->128 3x3 + bias + relu, WMMA im2col ----------
__global__ __launch_bounds__(256) void k_conv1(const unsigned short* __restrict__ X,
                                               const unsigned short* __restrict__ W1s,
                                               const float* __restrict__ b1,
                                               unsigned short* __restrict__ H1) {
  int wave = threadIdx.x >> 5, lane = threadIdx.x & 31;
  int hi = lane >> 4, l = lane & 15;
  int b = blockIdx.x / 72;
  int tIdx = (blockIdx.x % 72) * 8 + wave;           // 16-pixel tile within a row
  int row = tIdx / 6;
  int col = (tIdx % 6) * 16 + l;                     // per-lane pixel column
  v8f z = {};
  v8f acc[8];
#pragma unroll
  for (int m = 0; m < 8; ++m) acc[m] = z;
  for (int tap = 0; tap < 9; ++tap) {
    int ny = row + tap / 3 - 1;
    int nx = col + tap % 3 - 1;
    bool valid = ((unsigned)ny < 96u) && ((unsigned)nx < 96u);
    int np = (valid ? ny : 0) * 96 + (valid ? nx : 0);
    const unsigned short* bptr = X + ((size_t)b * HW_ + np) * CH1 + hi * 16;
    for (int ch = 0; ch < 9; ++ch) {
      v16u bf = {};
      if (valid) bf = *(const v16u*)(bptr + ch * 32);
      int kg = tap * CH1 + ch * 32;
#pragma unroll
      for (int m = 0; m < 8; ++m) {
        const unsigned short* arow = W1s + (size_t)(m * 16 + l) * K1;
        acc[m] = wmma_bf16(load_afrag(arow, kg, hi), bf, acc[m]);
      }
    }
  }
  unsigned short* hrow = H1 + ((size_t)b * HW_ + row * 96 + col) * 128;
#pragma unroll
  for (int m = 0; m < 8; ++m) {
    union { v8u v; unsigned short u[8]; } pk;
#pragma unroll
    for (int i = 0; i < 8; ++i) {
      int mm = m * 16 + hi * 8 + i;
      float v = acc[m][i] + b1[mm];
      pk.u[i] = f2bf(fmaxf(v, 0.f));
    }
    *(v8u*)(hrow + m * 16 + hi * 8) = pk.v;
  }
}

// ---------- 10. conv2: 128->2 3x3 + bias, WMMA (rows 2..15 of A are zero) ----------
__global__ __launch_bounds__(256) void k_conv2(const unsigned short* __restrict__ H1,
                                               const unsigned short* __restrict__ W2s,
                                               const float* __restrict__ b2,
                                               float* __restrict__ out) {
  int wave = threadIdx.x >> 5, lane = threadIdx.x & 31;
  int hi = lane >> 4, l = lane & 15;
  int b = blockIdx.x / 72;
  int tIdx = (blockIdx.x % 72) * 8 + wave;
  int row = tIdx / 6;
  int col = (tIdx % 6) * 16 + l;
  v8f acc = {};
  const unsigned short* arowBase = W2s + (size_t)l * K2;
  for (int tap = 0; tap < 9; ++tap) {
    int ny = row + tap / 3 - 1;
    int nx = col + tap % 3 - 1;
    bool valid = ((unsigned)ny < 96u) && ((unsigned)nx < 96u);
    int np = (valid ? ny : 0) * 96 + (valid ? nx : 0);
    const unsigned short* bptr = H1 + ((size_t)b * HW_ + np) * 128 + hi * 16;
#pragma unroll
    for (int ch = 0; ch < 4; ++ch) {
      v16u bf = {};
      if (valid) bf = *(const v16u*)(bptr + ch * 32);
      acc = wmma_bf16(load_afrag(arowBase, tap * 128 + ch * 32, hi), bf, acc);
    }
  }
  if (hi == 0) {
    int gp = row * 96 + col;
    out[((size_t)b * 2 + 0) * HW_ + gp] = acc[0] + b2[0];
    out[((size_t)b * 2 + 1) * HW_ + gp] = acc[1] + b2[1];
  }
}

extern "C" void kernel_launch(void* const* d_in, const int* in_sizes, int n_in,
                              void* d_out, int out_size, void* d_ws, size_t ws_size,
                              hipStream_t stream) {
  (void)in_sizes; (void)n_in; (void)out_size; (void)ws_size;
  const float* cur_embed  = (const float*)d_in[0];
  const float* cur_decode = (const float*)d_in[1];
  const float* init_embed = (const float*)d_in[2];
  const float* init_seg   = (const float*)d_in[3];
  const float* prev_embed = (const float*)d_in[4];
  const float* prev_seg   = (const float*)d_in[5];
  const float* W1 = (const float*)d_in[6];
  const float* b1 = (const float*)d_in[7];
  const float* W2 = (const float*)d_in[8];
  const float* b2 = (const float*)d_in[9];
  float* out = (float*)d_out;

  char* ws = (char*)d_ws;
  unsigned short* fi_t  = (unsigned short*)(ws + 0);          // 2,359,296 B
  float*          si_t  = (float*)(ws + 2359296);             //    36,864 B
  float*          gmap  = (float*)(ws + 2396160);             //   589,824 B
  float*          lmap  = (float*)(ws + 2985984);             //   589,824 B
  unsigned short* W1s   = (unsigned short*)(ws + 3575808);    //   663,552 B
  unsigned short* W2s   = (unsigned short*)(ws + 4239360);    //    36,864 B
  unsigned short* curT  = (unsigned short*)(ws + 4276224);    // 37,748,736 B
  unsigned short* prevT = (unsigned short*)(ws + 42024960);   // 37,748,736 B
  // Aliases (written only after the last reader of curT/prevT, stream-ordered):
  unsigned short* X     = (unsigned short*)(ws + 4276224);    // 42,467,328 B
  unsigned short* H1    = (unsigned short*)(ws + 46743552);   // 18,874,368 B

  k_pool_fi  <<<4608, 256, 0, stream>>>(init_embed, fi_t);
  k_pool_si  <<<18,   256, 0, stream>>>(init_seg, si_t);
  k_transpose<<<288,  256, 0, stream>>>(cur_embed, curT);
  k_transpose<<<288,  256, 0, stream>>>(prev_embed, prevT);
  k_w1prep   <<<1296, 256, 0, stream>>>(W1, W1s);
  k_w2prep   <<<72,   256, 0, stream>>>(W2, W2s);
  k_gmap     <<<576,  256, 0, stream>>>(fi_t, si_t, curT, gmap);
  k_corr     <<<1152, 256, 0, stream>>>(curT, prevT, prev_seg, lmap);
  k_stageX   <<<288,  256, 0, stream>>>(cur_decode, gmap, init_seg, lmap, prev_seg, X);
  k_conv1    <<<576,  256, 0, stream>>>(X, W1s, b1, H1);
  k_conv2    <<<576,  256, 0, stream>>>(H1, W2s, b2, out);
}
